// LSTMAgent_82317343195790
// MI455X (gfx1250) — matrix operand order repaired and checked by reading, not compiled, not run on hardware
//
#include <hip/hip_runtime.h>

typedef __attribute__((ext_vector_type(16))) __bf16 v16bf;
typedef __attribute__((ext_vector_type(8)))  float  v8f;

#define TSEQ   4096
#define HID    64
#define BTILE  16
#define CHUNK  64
#define NTHR   128

static __device__ __forceinline__ float fast_sigmoid(float x) {
    // 1 / (1 + 2^(-x*log2e)) : v_exp_f32 + v_rcp_f32
    return __builtin_amdgcn_rcpf(1.0f + __builtin_amdgcn_exp2f(-1.4426950408889634f * x));
}
static __device__ __forceinline__ float fast_tanh(float x) {
    return fmaf(2.0f, fast_sigmoid(2.0f * x), -1.0f);
}

__global__ __launch_bounds__(NTHR, 1)
void lstm_agent_kernel(const float* __restrict__ state,
                       const float* __restrict__ W_ih,
                       const float* __restrict__ W_hh,
                       const float* __restrict__ b_ih,
                       const float* __restrict__ b_hh,
                       const float* __restrict__ fc1_W,
                       const float* __restrict__ fc1_b,
                       const float* __restrict__ fc2_W,
                       const float* __restrict__ fc2_b,
                       float* __restrict__ out)
{
    const int tid   = threadIdx.x;
    const int wv    = tid >> 5;        // wave id 0..3
    const int lane  = tid & 31;
    const int l16   = lane & 15;
    const bool hiH  = lane >= 16;
    const int bbase = blockIdx.x * BTILE;

    __shared__ float  sbuf[BTILE][CHUNK * 2];   // staged state slice
    __shared__ __bf16 hbf[2][BTILE][HID];       // h ping-pong (bf16)
    __shared__ float  hf32[BTILE][HID];         // final h (fp32)
    __shared__ float  x1s[BTILE][HID];          // fc1 activations

    // Per-lane input-projection coefficients for the 4 gate tiles this wave owns.
    // Wave w owns columns w*16 + l16 of gate blocks j = 0..3 (i,f,g,o).
    float xw0[4], xw1[4], xb[4];
#pragma unroll
    for (int j = 0; j < 4; ++j) {
        const int g = j * HID + wv * 16 + l16;
        xw0[j] = W_ih[2 * g + 0];
        xw1[j] = W_ih[2 * g + 1];
        xb[j]  = b_ih[g] + b_hh[g];
    }

    // W_hh^T B-matrix fragments (bf16), resident in VGPRs for the whole scan.
    // B 16x16x32 layout: lane l16 = col N; lanes 0-15 hold K {0..7,16..23}, lanes 16-31 K {8..15,24..31}.
    // B[k][n] = W_hh[n][k] -> each lane reads two runs of 8 contiguous fp32 (16 apart) of a W_hh row.
    v16bf bfrag[4][2];
#pragma unroll
    for (int j = 0; j < 4; ++j) {
#pragma unroll
        for (int kb = 0; kb < 2; ++kb) {
            const int n    = j * HID + wv * 16 + l16;
            const int koff = kb * 32 + (hiH ? 8 : 0);
            const float* src = W_hh + n * HID + koff;
#pragma unroll
            for (int e = 0; e < 16; ++e) bfrag[j][kb][e] = (__bf16)src[(e & 7) + ((e >> 3) << 4)];
        }
    }

    // Register-resident cell state: lane holds c for (M = r + (hi?8:0), N = wv*16+l16).
    float cc[8];
#pragma unroll
    for (int r = 0; r < 8; ++r) cc[r] = 0.0f;

    for (int i = tid; i < BTILE * HID; i += NTHR)
        (&hbf[0][0][0])[i] = (__bf16)0.0f;
    __syncthreads();

    for (int t0 = 0; t0 < TSEQ; t0 += CHUNK) {
        // Stage 16 batches x CHUNK steps x 2 fp32 of state into LDS (float4 loads).
        for (int i = tid; i < (BTILE * CHUNK * 2) / 4; i += NTHR) {
            const int row = i >> 5;
            const int c4  = i & 31;
            const float4 v = *(const float4*)(state + (size_t)(bbase + row) * (TSEQ * 2)
                                              + t0 * 2 + c4 * 4);
            *(float4*)(&sbuf[row][c4 * 4]) = v;
        }
        if (t0 + CHUNK < TSEQ) {
            // CDNA5 global_prefetch_b8: pull next chunk toward the caches.
            __builtin_prefetch(state + (size_t)(bbase + (tid >> 3)) * (TSEQ * 2)
                               + (t0 + CHUNK) * 2 + (tid & 7) * 16, 0, 1);
        }
        __syncthreads();

        for (int tin = 0; tin < CHUNK; ++tin) {
            const int t = t0 + tin;
            const int p = t & 1;

            // acc init = x-projection (I=2 -> 2 FMAs per element) + biases
            v8f acc[4];
            float s0v[8], s1v[8];
#pragma unroll
            for (int r = 0; r < 8; ++r) {
                const int m = r + (hiH ? 8 : 0);
                const float2 s = *(const float2*)(&sbuf[m][tin * 2]);
                s0v[r] = s.x; s1v[r] = s.y;
            }
#pragma unroll
            for (int j = 0; j < 4; ++j)
#pragma unroll
                for (int r = 0; r < 8; ++r)
                    acc[j][r] = fmaf(s0v[r], xw0[j], fmaf(s1v[r], xw1[j], xb[j]));

            // A fragments: 16-bit A 16x32 layout; lanes 0-15: K {0..7,16..23},
            // lanes 16-31: K {8..15,24..31}. Two ds_load_b128 per fragment.
            v16bf afr[2];
#pragma unroll
            for (int kb = 0; kb < 2; ++kb) {
                const int e0 = kb * 32 + (hiH ? 8 : 0);
                union { v16bf v; uint4 q[2]; } u;
                u.q[0] = *(const uint4*)(&hbf[p][l16][e0]);
                u.q[1] = *(const uint4*)(&hbf[p][l16][e0 + 16]);
                afr[kb] = u.v;
            }

            // gates[16x16 tile j] += h[16x64] @ W_hh^T[64, tile j]  (K split in 2)
#pragma unroll
            for (int j = 0; j < 4; ++j)
                acc[j] = __builtin_amdgcn_wmma_f32_16x16x32_bf16(
                    false, afr[0], false, bfrag[j][0], (short)0, acc[j], false, false);
#pragma unroll
            for (int j = 0; j < 4; ++j)
                acc[j] = __builtin_amdgcn_wmma_f32_16x16x32_bf16(
                    false, afr[1], false, bfrag[j][1], (short)0, acc[j], false, false);

            // Elementwise LSTM cell update; wave owns full {i,f,g,o} for its 16 cols.
            const int col = wv * 16 + l16;
#pragma unroll
            for (int r = 0; r < 8; ++r) {
                const float iv = fast_sigmoid(acc[0][r]);
                const float fv = fast_sigmoid(acc[1][r]);
                const float gv = fast_tanh(acc[2][r]);
                const float ov = fast_sigmoid(acc[3][r]);
                cc[r] = fmaf(fv, cc[r], iv * gv);
                const float hv = ov * fast_tanh(cc[r]);
                const int m = r + (hiH ? 8 : 0);
                hbf[p ^ 1][m][col] = (__bf16)hv;
                if (t == TSEQ - 1) hf32[m][col] = hv;
            }
            __syncthreads();
        }
    }

    // fc1 + ReLU: 16 batches x 64 outputs over 128 threads (one-shot tail)
    {
        const int m  = tid & 15;
        const int og = tid >> 4;
#pragma unroll
        for (int oo = 0; oo < 8; ++oo) {
            const int o = og * 8 + oo;
            float a = fc1_b[o];
            for (int k = 0; k < HID; ++k)
                a = fmaf(hf32[m][k], fc1_W[o * HID + k], a);
            x1s[m][o] = fmaxf(a, 0.0f);
        }
    }
    __syncthreads();

    // fc2: 16 batches x 2 q-values
    if (tid < 32) {
        const int m = tid >> 1;
        const int q = tid & 1;
        float a = fc2_b[q];
        for (int k = 0; k < HID; ++k)
            a = fmaf(x1s[m][k], fc2_W[q * HID + k], a);
        out[(bbase + m) * 2 + q] = a;
    }
}

extern "C" void kernel_launch(void* const* d_in, const int* in_sizes, int n_in,
                              void* d_out, int out_size, void* d_ws, size_t ws_size,
                              hipStream_t stream) {
    (void)in_sizes; (void)n_in; (void)d_ws; (void)ws_size; (void)out_size;
    const float* state = (const float*)d_in[0];
    const float* W_ih  = (const float*)d_in[1];
    const float* W_hh  = (const float*)d_in[2];
    const float* b_ih  = (const float*)d_in[3];
    const float* b_hh  = (const float*)d_in[4];
    const float* fc1_W = (const float*)d_in[5];
    const float* fc1_b = (const float*)d_in[6];
    const float* fc2_W = (const float*)d_in[7];
    const float* fc2_b = (const float*)d_in[8];
    float* out = (float*)d_out;
    lstm_agent_kernel<<<dim3(256 / BTILE), dim3(NTHR), 0, stream>>>(
        state, W_ih, W_hh, b_ih, b_hh, fc1_W, fc1_b, fc2_W, fc2_b, out);
}
